// Decoder_38482906972240
// MI455X (gfx1250) — matrix operand order repaired and checked by reading, not compiled, not run on hardware
//
#include <hip/hip_runtime.h>
#include <hip/hip_bf16.h>
#include <stdint.h>

// ---------------------------------------------------------------------------
// 2-layer autoregressive LSTM decoder for MI455X (gfx1250), wave32 + WMMA.
//   B=1024, H=256, IN=64, OUT=64, T=512.  gates order: i,f,g,o (torch).
//
//   * Batch rows independent -> persistent kernel, 32 WGs x 32 rows, all 512
//     steps in-kernel, no global sync.
//   * Weights repacked once/launch into f16 WMMA B-fragments in d_ws
//     (~1.7 MB, L2-resident); f32 accumulate via v_wmma_f32_16x16x32_f16.
//   * Wave w owns hidden cols [w*32,w*32+32) of every gate block -> i/f/g/o
//     for a hidden unit meet in one wave; cell state c0/c1 stays in VGPRs.
//   * Opaque per-iteration integer offset blocks LICM weight-load hoisting
//     while preserving global address-space inference (global_load_b128).
//   * B fragments DOUBLE-BUFFERED across kt: 8 loads for kt+1 in flight while
//     the 8 WMMAs of kt execute (round-3 asm showed a single B register set
//     serializing load->wait->wmma per gate).
// ---------------------------------------------------------------------------

typedef __attribute__((ext_vector_type(16))) _Float16 v16h;
typedef __attribute__((ext_vector_type(8)))  float    v8f;

#define BATCH   1024
#define HID     256
#define INSZ    64
#define OUTSZ   64
#define TSTEPS  512
#define BT      32
#define NWG     (BATCH / BT)
#define XS      328          // xh0 LDS row stride (halfs):  x(64)|h0(256)+pad
#define HS      520          // hh1 LDS row stride (halfs): h0(256)|h1(256)+pad
#define PS      68           // pred LDS row stride (floats)

#define WS_L0_H   0
#define WS_L1_H   (64 * 10 * 512)
#define WS_FC_H   (WS_L1_H + 64 * 16 * 512)

// ---------------------------------------------------------------------------
// Weight repack: W[N x Ksrc] row-major (gates = act @ W^T, so B[k][n]=W[n][k])
// -> WMMA B-fragments (32x16 halfs). lane l: 16 halfs (32 B) contiguous,
// dword d -> K pair { (d<4?0:16)+(l>=16?8:0)+2*(d&3), +1 }, N = nt*16 + (l&15).
// ---------------------------------------------------------------------------
__global__ void repack_wmma_b(const float* __restrict__ W, _Float16* __restrict__ dst,
                              int ktiles, int kt_off, int KT_total, int Ksrc) {
  int frag = blockIdx.x;            // nt * ktiles + ktl
  int nt   = frag / ktiles;
  int ktl  = frag % ktiles;
  int lane = threadIdx.x & 31;
  int d    = threadIdx.x >> 5;      // 0..7
  int n    = nt * 16 + (lane & 15);
  int k    = ktl * 32 + ((lane >> 4) << 3) + ((d >> 2) << 4) + 2 * (d & 3);
  const float* src = W + (size_t)n * Ksrc + k;
  _Float16* o = dst + ((size_t)(nt * KT_total + kt_off + ktl)) * 512 + lane * 16 + d * 2;
  o[0] = (_Float16)src[0];
  o[1] = (_Float16)src[1];
}

// --------------------------- device helpers --------------------------------
__device__ __forceinline__ v8f wmma_f16(v16h a, v16h b, v8f c) {
  return __builtin_amdgcn_wmma_f32_16x16x32_f16(false, a, false, b, (short)0, c,
                                                false, false);
}
// WMMA with literal-zero accumulator (lowered to inline SRC2 = 0)
__device__ __forceinline__ v8f wmma_f16_z(v16h a, v16h b) {
  v8f z = {};
  return __builtin_amdgcn_wmma_f32_16x16x32_f16(false, a, false, b, (short)0, z,
                                                false, false);
}

// A fragment (16x32 f16) from LDS, ISA 16-bit A layout:
//   lane l: M = mrow0 + (l&15); dword j: K = (j<4?0:16) + (l>=16?8:0) + 2*(j&3)
// Bytes [0,16) and [32,48) per lane are contiguous -> ds_load_b128 pairs.
__device__ __forceinline__ v16h load_a_frag(const _Float16* buf, int stride,
                                            int mrow0, int kcol0, int lane) {
  const _Float16* rp = buf + (mrow0 + (lane & 15)) * stride + kcol0 + ((lane >> 4) << 3);
  union { v16h v; uint32_t u[8]; } r;
#pragma unroll
  for (int j = 0; j < 8; ++j) {
    int k = (j < 4) ? (2 * j) : (16 + 2 * (j - 4));
    r.u[j] = *(const uint32_t*)(rp + k);
  }
  return r.v;
}

__device__ __forceinline__ v16h load_b_frag(const _Float16* frag, int lane) {
  return *(const v16h*)(frag + lane * 16);   // 32 B/lane, coalesced 1 KB/wave
}

__device__ __forceinline__ float fsigmoid(float x) {
  return 1.0f / (1.0f + __expf(-x));
}
__device__ __forceinline__ float ftanh(float x) {
  float t = __expf(2.0f * x);
  return (t - 1.0f) / (t + 1.0f);
}

// One 16-column slice (nt) of the gate GEMM: acc[g][mt] over all 4 gates,
// both 16-row M tiles.  B fragments (global, L2-resident) double-buffered:
// the 8 loads for kt+1 are in flight while the 8 WMMAs of kt execute.
// kt=0 peeled (inline-0 accumulator), last kt peeled (no over-fetch).
__device__ __forceinline__ void gate_slice_gemm(const _Float16* A, int astride, int KT,
                                                const _Float16* B, int wave, int nt,
                                                int lane, v8f acc[4][2]) {
  const int ct0 = wave * 2 + nt;            // column tile of gate 0
  v16h bcur[4], bnxt[4];

  // ---- kt = 0 (peeled): C = inline 0; kt=1 loads issued before the WMMAs --
#pragma unroll
  for (int g = 0; g < 4; ++g)
    bcur[g] = load_b_frag(B + ((size_t)((g * 16 + ct0) * KT)) * 512, lane);
  {
    v16h a0 = load_a_frag(A, astride, 0,  0, lane);
    v16h a1 = load_a_frag(A, astride, 16, 0, lane);
#pragma unroll
    for (int g = 0; g < 4; ++g)
      bnxt[g] = load_b_frag(B + ((size_t)((g * 16 + ct0) * KT + 1)) * 512, lane);
#pragma unroll
    for (int g = 0; g < 4; ++g) {
      acc[g][0] = wmma_f16_z(a0, bcur[g]);
      acc[g][1] = wmma_f16_z(a1, bcur[g]);
    }
#pragma unroll
    for (int g = 0; g < 4; ++g) bcur[g] = bnxt[g];
  }

  // ---- steady state: branch-free, prefetch kt+1 while computing kt --------
#pragma unroll 2
  for (int kt = 1; kt < KT - 1; ++kt) {
#pragma unroll
    for (int g = 0; g < 4; ++g)
      bnxt[g] = load_b_frag(B + ((size_t)((g * 16 + ct0) * KT + kt + 1)) * 512, lane);
    v16h a0 = load_a_frag(A, astride, 0,  kt * 32, lane);
    v16h a1 = load_a_frag(A, astride, 16, kt * 32, lane);
#pragma unroll
    for (int g = 0; g < 4; ++g) {
      acc[g][0] = wmma_f16(a0, bcur[g], acc[g][0]);
      acc[g][1] = wmma_f16(a1, bcur[g], acc[g][1]);
    }
#pragma unroll
    for (int g = 0; g < 4; ++g) bcur[g] = bnxt[g];
  }

  // ---- kt = KT-1 (peeled): no further prefetch ----------------------------
  {
    v16h a0 = load_a_frag(A, astride, 0,  (KT - 1) * 32, lane);
    v16h a1 = load_a_frag(A, astride, 16, (KT - 1) * 32, lane);
#pragma unroll
    for (int g = 0; g < 4; ++g) {
      acc[g][0] = wmma_f16(a0, bcur[g], acc[g][0]);
      acc[g][1] = wmma_f16(a1, bcur[g], acc[g][1]);
    }
  }
}

// ---------------------------------------------------------------------------
// Persistent decoder kernel: 32 blocks x 256 threads (8 waves).
// ---------------------------------------------------------------------------
__global__ void __launch_bounds__(256)
lstm_decoder_kernel(const float* __restrict__ h0_in, const float* __restrict__ c0_in,
                    const float* __restrict__ bih0, const float* __restrict__ bhh0,
                    const float* __restrict__ bih1, const float* __restrict__ bhh1,
                    const float* __restrict__ bfc,
                    const _Float16* __restrict__ wsL0_in,
                    const _Float16* __restrict__ wsL1_in,
                    const _Float16* __restrict__ wsFC_in,
                    float* __restrict__ out) {
  __shared__ _Float16 xh0[BT * XS];   // [32][64 x | 256 h0]
  __shared__ _Float16 hh1[BT * HS];   // [32][256 h0 | 256 h1]
  __shared__ float    pred[BT * PS];  // [32][64] fc output

  const int tid  = threadIdx.x;
  const int wave = tid >> 5;
  const int lane = tid & 31;
  const int brow = blockIdx.x * BT;
  const int hi8  = (lane >> 4) << 3;        // C-matrix row offset (0 or 8)
  const int ncol = wave * 32 + (lane & 15); // + nt*16 -> this lane's hidden col

  // ---- init x (SOS one-hot at col 61) and h state in LDS -------------------
  for (int i = tid; i < BT * INSZ; i += 256) {
    int r = i >> 6, c = i & 63;
    xh0[r * XS + c] = (c == INSZ - 3) ? (_Float16)1.0f : (_Float16)0.0f;
  }
  for (int i = tid; i < BT * HID; i += 256) {
    int r = i >> 8, c = i & 255;
    float h0v = h0_in[(size_t)(brow + r) * HID + c];
    float h1v = h0_in[(size_t)BATCH * HID + (size_t)(brow + r) * HID + c];
    _Float16 h0h = (_Float16)h0v;
    xh0[r * XS + INSZ + c] = h0h;
    hh1[r * HS + c]        = h0h;
    hh1[r * HS + HID + c]  = (_Float16)h1v;
  }

  // ---- cell state in registers for all 512 steps ---------------------------
  v8f c0r[2][2], c1r[2][2];                 // [mt][nt]
#pragma unroll
  for (int mt = 0; mt < 2; ++mt)
#pragma unroll
    for (int nt = 0; nt < 2; ++nt)
#pragma unroll
      for (int r = 0; r < 8; ++r) {
        int m   = mt * 16 + r + hi8;
        int col = ncol + nt * 16;
        c0r[mt][nt][r] = c0_in[(size_t)(brow + m) * HID + col];
        c1r[mt][nt][r] = c0_in[(size_t)BATCH * HID + (size_t)(brow + m) * HID + col];
      }
  __syncthreads();

  v8f acc[4][2];     // [gate][mt] for current nt slice
  v8f hreg[2][2];    // [mt][nt]   deferred h_new stores

  for (int t = 0; t < TSTEPS; ++t) {
    // Opaque integer offset: blocks LICM hoisting of the loop-invariant
    // weight loads but keeps address-space inference -> global_load_b128.
    size_t woff = 0;
    asm volatile("" : "+s"(woff));
    const _Float16* wsL0 = wsL0_in + woff;
    const _Float16* wsL1 = wsL1_in + woff;
    const _Float16* wsFC = wsFC_in + woff;

    // ================= layer 0: gates = [x|h0] @ W^T, K = 320 ==============
#pragma unroll
    for (int nt = 0; nt < 2; ++nt) {
      gate_slice_gemm(xh0, XS, 10, wsL0, wave, nt, lane, acc);
      int col = ncol + nt * 16;
#pragma unroll
      for (int g = 0; g < 4; ++g) {
        float bv = bih0[g * 256 + col] + bhh0[g * 256 + col];
#pragma unroll
        for (int mt = 0; mt < 2; ++mt)
#pragma unroll
          for (int r = 0; r < 8; ++r) acc[g][mt][r] += bv;
      }
#pragma unroll
      for (int mt = 0; mt < 2; ++mt)
#pragma unroll
        for (int r = 0; r < 8; ++r) {
          float cn = fsigmoid(acc[1][mt][r]) * c0r[mt][nt][r] +
                     fsigmoid(acc[0][mt][r]) * ftanh(acc[2][mt][r]);
          hreg[mt][nt][r] = fsigmoid(acc[3][mt][r]) * ftanh(cn);
          c0r[mt][nt][r]  = cn;
        }
    }
    __syncthreads();   // all waves done reading xh0 before h0 is overwritten
#pragma unroll
    for (int mt = 0; mt < 2; ++mt)
#pragma unroll
      for (int nt = 0; nt < 2; ++nt)
#pragma unroll
        for (int r = 0; r < 8; ++r) {
          int m   = mt * 16 + r + hi8;
          int col = ncol + nt * 16;
          _Float16 hh = (_Float16)hreg[mt][nt][r];
          xh0[m * XS + INSZ + col] = hh;   // h0 state for next step
          hh1[m * HS + col]        = hh;   // h0_new feeding layer 1 now
        }
    __syncthreads();

    // ================= layer 1: gates = [h0_new|h1] @ W^T, K = 512 =========
#pragma unroll
    for (int nt = 0; nt < 2; ++nt) {
      gate_slice_gemm(hh1, HS, 16, wsL1, wave, nt, lane, acc);
      int col = ncol + nt * 16;
#pragma unroll
      for (int g = 0; g < 4; ++g) {
        float bv = bih1[g * 256 + col] + bhh1[g * 256 + col];
#pragma unroll
        for (int mt = 0; mt < 2; ++mt)
#pragma unroll
          for (int r = 0; r < 8; ++r) acc[g][mt][r] += bv;
      }
#pragma unroll
      for (int mt = 0; mt < 2; ++mt)
#pragma unroll
        for (int r = 0; r < 8; ++r) {
          float cn = fsigmoid(acc[1][mt][r]) * c1r[mt][nt][r] +
                     fsigmoid(acc[0][mt][r]) * ftanh(acc[2][mt][r]);
          hreg[mt][nt][r] = fsigmoid(acc[3][mt][r]) * ftanh(cn);
          c1r[mt][nt][r]  = cn;
        }
    }
    __syncthreads();   // all waves done reading hh1 before h1 is overwritten
#pragma unroll
    for (int mt = 0; mt < 2; ++mt)
#pragma unroll
      for (int nt = 0; nt < 2; ++nt)
#pragma unroll
        for (int r = 0; r < 8; ++r) {
          int m   = mt * 16 + r + hi8;
          int col = ncol + nt * 16;
          hh1[m * HS + HID + col] = (_Float16)hreg[mt][nt][r]; // h1_new
        }
    __syncthreads();

    // ================= FC: pred[32x64] = h1_new @ Wfc^T + bfc ==============
    {
      int mt  = wave >> 2;
      int ntf = wave & 3;
      v8f pacc;
      {
        v16h a = load_a_frag(hh1, HS, mt * 16, HID, lane);
        v16h b = load_b_frag(wsFC + ((size_t)(ntf * 8)) * 512, lane);
        pacc = wmma_f16_z(a, b);
      }
#pragma unroll 1
      for (int kt = 1; kt < 8; ++kt) {
        v16h a = load_a_frag(hh1, HS, mt * 16, HID + kt * 32, lane);
        v16h b = load_b_frag(wsFC + ((size_t)(ntf * 8 + kt)) * 512, lane);
        pacc = wmma_f16(a, b, pacc);
      }
      float bv = bfc[ntf * 16 + (lane & 15)];
#pragma unroll
      for (int r = 0; r < 8; ++r)
        pred[(mt * 16 + r + hi8) * PS + ntf * 16 + (lane & 15)] = pacc[r] + bv;
    }
    __syncthreads();

    // ====== output head: log_softmax(cols 0..62) | sigmoid(col 63) =========
    {
      int orow = tid >> 3;        // 8 threads per batch row
      int cg   = tid & 7;         // each covers 8 columns
      const float* pr = pred + orow * PS;
      float mx = -1e30f;
#pragma unroll
      for (int j = 0; j < 8; ++j) {
        int c = cg * 8 + j;
        if (c < OUTSZ - 1) mx = fmaxf(mx, pr[c]);
      }
      mx = fmaxf(mx, __shfl_xor(mx, 1, 8));
      mx = fmaxf(mx, __shfl_xor(mx, 2, 8));
      mx = fmaxf(mx, __shfl_xor(mx, 4, 8));
      float s = 0.0f;
#pragma unroll
      for (int j = 0; j < 8; ++j) {
        int c = cg * 8 + j;
        if (c < OUTSZ - 1) s += __expf(pr[c] - mx);
      }
      s += __shfl_xor(s, 1, 8);
      s += __shfl_xor(s, 2, 8);
      s += __shfl_xor(s, 4, 8);
      float logz = mx + __logf(s);
      float* orow_out = out + ((size_t)(brow + orow) * TSTEPS + t) * OUTSZ;
#pragma unroll
      for (int j = 0; j < 8; ++j) {
        int c = cg * 8 + j;
        float pv = pr[c];
        float ov = (c == OUTSZ - 1) ? fsigmoid(pv) : (pv - logz);
        orow_out[c] = ov;                  // stream result to HBM
        xh0[orow * XS + c] = (_Float16)ov; // autoregressive: next x = out
      }
    }
    __syncthreads();
  }

  // ---- final states: h_f (from LDS f16), c_f (from registers) --------------
  const size_t HF = (size_t)BATCH * TSTEPS * OUTSZ;
  const size_t CF = HF + (size_t)2 * BATCH * HID;
  for (int i = tid; i < BT * HID; i += 256) {
    int r = i >> 8, c = i & 255;
    out[HF + (size_t)(brow + r) * HID + c] = (float)hh1[r * HS + c];
    out[HF + (size_t)BATCH * HID + (size_t)(brow + r) * HID + c] =
        (float)hh1[r * HS + HID + c];
  }
#pragma unroll
  for (int mt = 0; mt < 2; ++mt)
#pragma unroll
    for (int nt = 0; nt < 2; ++nt)
#pragma unroll
      for (int r = 0; r < 8; ++r) {
        int m   = mt * 16 + r + hi8;
        int col = ncol + nt * 16;
        out[CF + (size_t)(brow + m) * HID + col] = c0r[mt][nt][r];
        out[CF + (size_t)BATCH * HID + (size_t)(brow + m) * HID + col] = c1r[mt][nt][r];
      }
}

// ---------------------------------------------------------------------------
extern "C" void kernel_launch(void* const* d_in, const int* in_sizes, int n_in,
                              void* d_out, int out_size, void* d_ws, size_t ws_size,
                              hipStream_t stream) {
  (void)in_sizes; (void)n_in; (void)out_size; (void)ws_size;
  const float* h0   = (const float*)d_in[1];
  const float* c0   = (const float*)d_in[2];
  const float* Wih0 = (const float*)d_in[3];
  const float* Whh0 = (const float*)d_in[4];
  const float* bih0 = (const float*)d_in[5];
  const float* bhh0 = (const float*)d_in[6];
  const float* Wih1 = (const float*)d_in[7];
  const float* Whh1 = (const float*)d_in[8];
  const float* bih1 = (const float*)d_in[9];
  const float* bhh1 = (const float*)d_in[10];
  const float* Wfc  = (const float*)d_in[11];
  const float* bfc  = (const float*)d_in[12];

  _Float16* ws   = (_Float16*)d_ws;          // needs ~1.7 MB of workspace
  _Float16* wsL0 = ws + WS_L0_H;
  _Float16* wsL1 = ws + WS_L1_H;
  _Float16* wsFC = ws + WS_FC_H;

  // One-time (per launch) weight repack into WMMA B-fragments, f32 -> f16.
  repack_wmma_b<<<64 * 2, 256, 0, stream>>>(Wih0, wsL0, 2, 0, 10, INSZ);
  repack_wmma_b<<<64 * 8, 256, 0, stream>>>(Whh0, wsL0, 8, 2, 10, HID);
  repack_wmma_b<<<64 * 8, 256, 0, stream>>>(Wih1, wsL1, 8, 0, 16, HID);
  repack_wmma_b<<<64 * 8, 256, 0, stream>>>(Whh1, wsL1, 8, 8, 16, HID);
  repack_wmma_b<<< 4 * 8, 256, 0, stream>>>(Wfc,  wsFC, 8, 0,  8, HID);

  // Persistent decoder: all 512 timesteps in one kernel.
  lstm_decoder_kernel<<<NWG, 256, 0, stream>>>(h0, c0, bih0, bhh0, bih1, bhh1,
                                               bfc, wsL0, wsL1, wsFC,
                                               (float*)d_out);
}
